// Amortized_VI_41360535060907
// MI455X (gfx1250) — compile-verified, matrix operand order checked by reading, not run on hardware
//
#include <hip/hip_runtime.h>
#include <hip/hip_bf16.h>

typedef _Float16 v16h  __attribute__((ext_vector_type(16)));
typedef float    v8f   __attribute__((ext_vector_type(8)));
typedef float    f32x4 __attribute__((ext_vector_type(4)));

union FragU { f32x4 q[2]; v16h h; };

#define LOG2PI_F 1.8378770664093453f
#define NOISE_F  0.015f

// ---- weight image layout (identical in workspace and LDS) ----
// hidden B-frags : 15 layers * 6 blocks(3 ntile x 2 kchunk) * 1024B
#define WHID_OFF   0
// input-layer B-frags : 3 blocks * 1024B
#define WINP_OFF   92160
// biases: 16 rows (inp + 15 hidden) * 48 f32
#define BIAS_OFF   95232
// head: 40 w + 1 b + pad, f32
#define OUTW_OFF   98304
#define WREG_BYTES 98560              // 16B-aligned copy size
#define H_OFF      WREG_BYTES         // LDS: 8 waves * 16x64 f16 tiles
#define SMEM_BYTES (WREG_BYTES + 8*2048)

struct PinnP {
    const float* Wh[15]; const float* bh[15];
    const float* Wi; const float* bi;
    const float* Wo; const float* bo;
};
struct EncP { const float* W[15]; const float* B[15]; };

// fast silu: x * rcp(1+exp(-x)) — v_exp_f32 + v_rcp_f32 are TRANS ops that
// co-execute with the XDL/WMMA pipe; avoids the IEEE v_div_scale chain.
__device__ __forceinline__ float fast_silu(float x) {
    return x * __builtin_amdgcn_rcpf(1.f + __expf(-x));
}

// fragment element mapping (mirrors the 16-bit A layout with M<->N swapped):
// lane = n + 16*half ; vgpr v, half-of-dword hi -> k = (v&3)*2 + (v>>2)*16 + half*8 + hi
__device__ __forceinline__ int frag_k(int v, int hi, int half) {
    return (v & 3) * 2 + ((v >> 2) * 16) + half * 8 + hi;
}

// =====================  weight prep + d_out zero  =====================
__global__ __launch_bounds__(256) void prep_kernel(PinnP p, char* ws, float* out0)
{
    int t = blockIdx.x * blockDim.x + threadIdx.x;
    if (t == 0) *out0 = 0.f;

    if (t < 2880) {                              // hidden-layer B fragments
        int lh = t / 192, rem = t % 192, blk = rem / 32, lane = rem % 32;
        int ntile = blk >> 1, kch = blk & 1;
        int nloc = lane & 15, half = lane >> 4;
        _Float16* dst = (_Float16*)(ws + (size_t)(lh * 6 + blk) * 1024 + lane * 32);
        const float* W = p.Wh[lh];               // (40,40) row-major (K,N)
        for (int v = 0; v < 8; ++v)
            for (int hi = 0; hi < 2; ++hi) {
                int gk = kch * 32 + frag_k(v, hi, half);
                int gn = ntile * 16 + nloc;
                float val = (gk < 40 && gn < 40) ? W[gk * 40 + gn] : 0.f;
                dst[v * 2 + hi] = (_Float16)val;
            }
    } else if (t < 2976) {                       // input-layer B fragments (K pad 6->32)
        int t2 = t - 2880, ntile = t2 / 32, lane = t2 % 32;
        int nloc = lane & 15, half = lane >> 4;
        _Float16* dst = (_Float16*)(ws + WINP_OFF + (size_t)ntile * 1024 + lane * 32);
        const float* W = p.Wi;                   // (6,40)
        for (int v = 0; v < 8; ++v)
            for (int hi = 0; hi < 2; ++hi) {
                int gk = frag_k(v, hi, half);
                int gn = ntile * 16 + nloc;
                float val = (gk < 6 && gn < 40) ? W[gk * 40 + gn] : 0.f;
                dst[v * 2 + hi] = (_Float16)val;
            }
    } else if (t < 2976 + 768) {                 // biases (padded to 48)
        int t3 = t - 2976, lr = t3 / 48, col = t3 % 48;
        float v = 0.f;
        if (col < 40) v = (lr == 0) ? p.bi[col] : p.bh[lr - 1][col];
        ((float*)(ws + BIAS_OFF))[lr * 48 + col] = v;
    } else if (t < 2976 + 768 + 44) {            // head
        int t4 = t - 3744;
        float v = 0.f;
        if (t4 < 40) v = p.Wo[t4];
        else if (t4 == 40) v = p.bo[0];
        ((float*)(ws + OUTW_OFF))[t4] = v;
    }
}

// =====================  encoder: 3 MLPs + L, xi, prior, entropy  =====================
// one wave per sample; intra-wave LDS traffic only (DS is in-order per wave)
__global__ __launch_bounds__(256) void enc_kernel(const float* y, const float* z, EncP p,
                                                  float* xi, float* lp, float* ent,
                                                  int n, int P)
{
    __shared__ float buf[8][2][64];
    __shared__ float res[8][20];
    __shared__ float xib[8][40];
    int w = threadIdx.x >> 5, l = threadIdx.x & 31;
    int i = blockIdx.x * 8 + w;
    if (i >= n) return;

    const int din[5] = {9, 50, 40, 30, 20};
    float* A  = buf[w][0];
    float* Bv = buf[w][1];

    for (int m = 0; m < 3; ++m) {
        if (l < 9) A[l] = y[i * 9 + l];
        asm volatile("" ::: "memory");
        for (int L = 0; L < 5; ++L) {
            int dk = din[L];
            int dn = (L < 4) ? din[L + 1] : ((m == 2) ? 10 : 5);
            const float* W  = p.W[m * 5 + L];
            const float* Bb = p.B[m * 5 + L];
            int j0 = l, j1 = l + 32;
            float a0 = 0.f, a1 = 0.f;
            for (int k = 0; k < dk; ++k) {
                float hv = A[k];
                if (j0 < dn) a0 += hv * W[k * dn + j0];
                if (j1 < dn) a1 += hv * W[k * dn + j1];
            }
            if (j0 < dn) { a0 += Bb[j0]; if (L < 4) a0 = fmaxf(a0, 0.f); Bv[j0] = a0; }
            if (j1 < dn) { a1 += Bb[j1]; if (L < 4) a1 = fmaxf(a1, 0.f); Bv[j1] = a1; }
            asm volatile("" ::: "memory");
            float* tmp = A; A = Bv; Bv = tmp;
        }
        int dlast = (m == 2) ? 10 : 5;
        if (l < dlast) res[w][m * 5 + l] = A[l];    // mu:0-4  ldraw:5-9  loff:10-19
        asm volatile("" ::: "memory");
    }

    const int obase[5] = {0, 0, 1, 3, 6};           // tril(-1) row starts
    if (l < 40) {
        int pp = l / 5, d = l % 5;
        float ldr = res[w][5 + d];
        float Ld  = (ldr > 20.f) ? ldr : log1pf(__expf(ldr));   // softplus
        float v   = res[w][d];
        const float* zr = z + (size_t)(i * P + pp) * 5;
        for (int j = 0; j < d; ++j) v += res[w][10 + obase[d] + j] * zr[j];
        v += Ld * zr[d];
        xi[(size_t)(i * P + pp) * 5 + d] = v;
        xib[w][l] = v;
    }
    asm volatile("" ::: "memory");
    if (l < P) {
        float s2 = 0.f;
        for (int d = 0; d < 5; ++d) { float u = xib[w][l * 5 + d]; s2 += u * u; }
        lp[(size_t)i * P + l] = -0.5f * s2 - 2.5f * LOG2PI_F;
    }
    if (l == 0) {
        float e = 0.5f * 5.f * (1.f + LOG2PI_F);
        for (int d = 0; d < 5; ++d) {
            float ldr = res[w][5 + d];
            float Ld  = (ldr > 20.f) ? ldr : log1pf(__expf(ldr));
            e += logf(Ld);
        }
        ent[i] = e;
    }
}

// =====================  PINN: WMMA pipeline  =====================
__global__ __launch_bounds__(256) void pinn_kernel(const f32x4* wsrc, const float* xi,
                                                   float* frow, int rows)
{
    extern __shared__ __align__(16) char smem[];
    // hint GL2 to pull the weight image early (global_prefetch_b8)
    __builtin_prefetch((const char*)wsrc + (threadIdx.x << 8), 0, 1);
    {   // stage the whole weight image into LDS once per workgroup
        f32x4* d = (f32x4*)smem;
        for (int idx = threadIdx.x; idx < WREG_BYTES / 16; idx += 256) d[idx] = wsrc[idx];
    }
    __syncthreads();

    int  w = threadIdx.x >> 5, lane = threadIdx.x & 31;
    int  nl = lane & 15, half = lane >> 4;
    int  tile = blockIdx.x * 8 + w;
    int  rowBase = tile * 16;
    char* hb = smem + H_OFF + w * 2048;                 // 16 rows x 64 f16, pitch 128B
    const float* biasL = (const float*)(smem + BIAS_OFF);
    const float* outw  = (const float*)(smem + OUTW_OFF);

    // zero activation panel, then fill padded inputs [x_tick, xi(5), 0...]
    {
        f32x4 zz = {0.f, 0.f, 0.f, 0.f};
        f32x4* zp = (f32x4*)(hb + lane * 64);
        zp[0] = zz; zp[1] = zz; zp[2] = zz; zp[3] = zz;
    }
    asm volatile("" ::: "memory");
    if (lane < 16) {
        int row = rowBase + lane;
        if (row < rows) {
            int s = row / 9, t = row % 9;
            _Float16* hr = (_Float16*)(hb + lane * 128);
            hr[0] = (_Float16)(0.15f + 0.0875f * (float)t);
            const float* xr = xi + (size_t)s * 5;
            for (int d = 0; d < 5; ++d) hr[1 + d] = (_Float16)xr[d];
        }
    }
    asm volatile("" ::: "memory");

    int abase = nl * 128 + half * 16;
    auto loadA = [&](int kch) -> v16h {
        FragU f;
        const char* pA = hb + abase + kch * 64;
        f.q[0] = *(const f32x4*)pA;
        f.q[1] = *(const f32x4*)(pA + 32);
        return f.h;
    };
    auto loadB = [&](int blk) -> v16h {
        FragU f;
        const char* pB = smem + (size_t)blk * 1024 + lane * 32;
        f.q[0] = *(const f32x4*)pB;
        f.q[1] = *(const f32x4*)(pB + 16);
        return f.h;
    };
    auto storeC = [&](v8f c0, v8f c1, v8f c2) {
        asm volatile("" ::: "memory");
        int mb = half * 8;
        #pragma unroll
        for (int r = 0; r < 8; ++r) {
            *(_Float16*)(hb + (mb + r) * 128 + (0 * 16 + nl) * 2) = (_Float16)c0[r];
            *(_Float16*)(hb + (mb + r) * 128 + (1 * 16 + nl) * 2) = (_Float16)c1[r];
            *(_Float16*)(hb + (mb + r) * 128 + (2 * 16 + nl) * 2) = (_Float16)c2[r];
        }
        asm volatile("" ::: "memory");
    };

    v8f h0, h1, h2;                                      // residual carry (C layout, f32)

    // ---- input layer: 16x32 x 32x48 ----
    {
        v16h a0 = loadA(0);
        v8f c[3];
        #pragma unroll
        for (int nt = 0; nt < 3; ++nt) {
            v8f acc = {0.f, 0.f, 0.f, 0.f, 0.f, 0.f, 0.f, 0.f};
            v16h bm = loadB(WINP_OFF / 1024 + nt);
            acc = __builtin_amdgcn_wmma_f32_16x16x32_f16(false, a0, false, bm,
                                                         (short)0, acc, false, false);
            float bv = biasL[0 * 48 + nt * 16 + nl];
            #pragma unroll
            for (int r = 0; r < 8; ++r) acc[r] = fast_silu(acc[r] + bv);
            c[nt] = acc;
        }
        h0 = c[0]; h1 = c[1]; h2 = c[2];
        storeC(c[0], c[1], c[2]);
    }

    // ---- 5 residual blocks x 3 layers: 16x64 x 64x48 each (fully unrolled) ----
    #pragma unroll
    for (int b = 0; b < 5; ++b) {
        #pragma unroll
        for (int j = 0; j < 3; ++j) {
            int lh = b * 3 + j;
            v16h a0 = loadA(0), a1 = loadA(1);
            v8f c[3];
            #pragma unroll
            for (int nt = 0; nt < 3; ++nt) {
                v8f acc = {0.f, 0.f, 0.f, 0.f, 0.f, 0.f, 0.f, 0.f};
                v16h b0 = loadB(lh * 6 + nt * 2 + 0);
                acc = __builtin_amdgcn_wmma_f32_16x16x32_f16(false, a0, false, b0,
                                                             (short)0, acc, false, false);
                v16h b1 = loadB(lh * 6 + nt * 2 + 1);
                acc = __builtin_amdgcn_wmma_f32_16x16x32_f16(false, a1, false, b1,
                                                             (short)0, acc, false, false);
                float bv = biasL[(1 + lh) * 48 + nt * 16 + nl];
                #pragma unroll
                for (int r = 0; r < 8; ++r) acc[r] = fast_silu(acc[r] + bv);
                c[nt] = acc;
            }
            if (j == 2) {                                // h = h + r
                c[0] += h0; c[1] += h1; c[2] += h2;
                h0 = c[0]; h1 = c[1]; h2 = c[2];
            }
            if (!(b == 4 && j == 2)) storeC(c[0], c[1], c[2]);
        }
    }

    // ---- head: 40 -> 1 via lane reduction over C layout ----
    float pr[8];
    #pragma unroll
    for (int r = 0; r < 8; ++r)
        pr[r] = h0[r] * outw[nl] + h1[r] * outw[16 + nl] + h2[r] * outw[32 + nl];
    #pragma unroll
    for (int m = 1; m < 16; m <<= 1)
        #pragma unroll
        for (int r = 0; r < 8; ++r) pr[r] += __shfl_xor(pr[r], m, 16);
    if (nl < 8) {
        int row = rowBase + half * 8 + nl;
        if (row < rows) {
            int t = row % 9;
            float x = 0.15f + 0.0875f * (float)t;
            float net = pr[nl] + outw[40];
            frow[row] = (1.f - x) + x * (1.f - x) * net;   // B1=1, B2=0
        }
    }
}

// =====================  final reduction  =====================
__global__ __launch_bounds__(256) void reduce_kernel(const float* y, const float* frow,
                                                     const float* lp, const float* ent,
                                                     float* out, int n, int P)
{
    int s = blockIdx.x * blockDim.x + threadIdx.x;
    int nP = n * P;
    if (s >= nP) return;
    int i = s / P;
    float acc = 0.f;
    float inv = 1.f / NOISE_F;
    for (int t = 0; t < 9; ++t) {
        float r = (y[i * 9 + t] - frow[s * 9 + t]) * inv;
        acc += -0.5f * r * r;
    }
    acc += 9.f * (-logf(NOISE_F) - 0.5f * LOG2PI_F);
    float val = lp[s] + acc;
    atomicAdd(out, val / (float)nP);
    if ((s % P) == 0) atomicAdd(out, ent[i] / (float)n);
}

// =====================  launch  =====================
extern "C" void kernel_launch(void* const* d_in, const int* in_sizes, int n_in,
                              void* d_out, int out_size, void* d_ws, size_t ws_size,
                              hipStream_t stream)
{
    const float* y = (const float*)d_in[0];
    const float* z = (const float*)d_in[1];
    int n  = in_sizes[0] / 9;
    int P  = in_sizes[1] / (n * 5);
    int nP = n * P;
    int rows = nP * 9;

    EncP ep;
    for (int m = 0; m < 3; ++m)
        for (int L = 0; L < 5; ++L) {
            ep.W[m * 5 + L] = (const float*)d_in[2 + m * 10 + 2 * L];
            ep.B[m * 5 + L] = (const float*)d_in[2 + m * 10 + 2 * L + 1];
        }
    PinnP pp;
    for (int l = 0; l < 15; ++l) {                 // pinn dict order: blocks, inp, out
        pp.Wh[l] = (const float*)d_in[32 + 2 * l];
        pp.bh[l] = (const float*)d_in[33 + 2 * l];
    }
    pp.Wi = (const float*)d_in[62]; pp.bi = (const float*)d_in[63];
    pp.Wo = (const float*)d_in[64]; pp.bo = (const float*)d_in[65];

    char*  ws   = (char*)d_ws;
    float* xi   = (float*)(ws + WREG_BYTES);
    float* lp   = xi + (size_t)nP * 5;
    float* ent  = lp + nP;
    float* frow = ent + n;

    prep_kernel<<<15, 256, 0, stream>>>(pp, ws, (float*)d_out);
    enc_kernel<<<(n + 7) / 8, 256, 0, stream>>>(y, z, ep, xi, lp, ent, n, P);
    int tiles = (rows + 15) / 16;
    int blocks = (tiles + 7) / 8;
    pinn_kernel<<<blocks, 256, SMEM_BYTES, stream>>>((const f32x4*)ws, xi, frow, rows);
    reduce_kernel<<<(nP + 255) / 256, 256, 0, stream>>>(y, frow, lp, ent, (float*)d_out, n, P);
}
